// SModule_80367428043062
// MI455X (gfx1250) — matrix operand order repaired and checked by reading, not compiled, not run on hardware
//
#include <hip/hip_runtime.h>
#include <hip/hip_bf16.h>

// ---- problem constants -----------------------------------------------------
#define BB      8
#define DD      128
#define NN      2048
#define LL      12
#define HH      8
#define DKK     16
#define NLP     24576      // N*L (pixels per batch)
#define BSTRIDE 3145728    // D*N*L
#define PTILES  768        // NLP / 32
#define TOT     25165824   // B*D*N*L
#define GRP     96         // B*L  (layernorm groups)
#define GRPSZ   262144     // D*N  (elements per group)

typedef __attribute__((ext_vector_type(16))) __bf16 bf16x16;
typedef __attribute__((ext_vector_type(8)))  __bf16 bf16x8;
typedef __attribute__((ext_vector_type(4)))  __bf16 bf16x4;
typedef __attribute__((ext_vector_type(8)))  float  f32x8;

__device__ __forceinline__ bf16x8 ld8(const __bf16* p) { return *(const bf16x8*)p; }

__device__ __forceinline__ bf16x16 cat8(bf16x8 lo, bf16x8 hi) {
    return __builtin_shufflevector(lo, hi, 0,1,2,3,4,5,6,7,8,9,10,11,12,13,14,15);
}

__device__ __forceinline__ f32x8 wmma_bf16(bf16x16 a, bf16x16 b, f32x8 c) {
    // (neg_a, A, neg_b, B, c_mod, C, reuse_a, reuse_b)
    return __builtin_amdgcn_wmma_f32_16x16x32_bf16(false, a, false, b, (short)0, c, false, false);
}

__device__ __forceinline__ f32x8 zero8() { f32x8 z = {0,0,0,0,0,0,0,0}; return z; }

__device__ __forceinline__ float sigmoidf_(float x) { return 1.0f / (1.0f + __expf(-x)); }

// ---- tiny prep kernels -----------------------------------------------------
__global__ void cvt_bf16_kernel(const float* __restrict__ src, __bf16* __restrict__ dst, int n) {
    int i = blockIdx.x * blockDim.x + threadIdx.x;
    if (i < n) dst[i] = (__bf16)src[i];
}

// key_sm = softmax(s_attn * 0.25 over DK), stored transposed bf16: KT[h][l][x][n]
__global__ void key_softmax_kernel(const float* __restrict__ s_attn, __bf16* __restrict__ KT) {
    int i = blockIdx.x * blockDim.x + threadIdx.x;  // over H*L*N
    if (i >= HH * LL * NN) return;
    int n  = i % NN;
    int hl = i / NN;
    const float* src = s_attn + (size_t)i * DKK;
    float v[DKK];
    float mx = -3.4e38f;
    #pragma unroll
    for (int x = 0; x < DKK; ++x) { v[x] = src[x] * 0.25f; mx = fmaxf(mx, v[x]); }
    float s = 0.f;
    #pragma unroll
    for (int x = 0; x < DKK; ++x) { v[x] = __expf(v[x] - mx); s += v[x]; }
    float inv = 1.0f / s;
    __bf16* dst = KT + (size_t)hl * (DKK * NN) + n;
    #pragma unroll
    for (int x = 0; x < DKK; ++x) dst[(size_t)x * NN] = (__bf16)(v[x] * inv);
}

// ---- generic WMMA conv1x1:  Y = act(W @ X + b) [+ resid] -------------------
// X,Y layout (B,128,N,L); block = 256 thr = 8 waves; tile O=128 x P=32.
#define CONV_MAINLOOP(ACCLIST)                                                          \
    for (int k0 = 0; k0 < DD; k0 += 32) {                                               \
        __syncthreads();                                                                \
        {                                                                               \
            const float* src = xb + (size_t)(k0 + skk) * NLP + p0 + spp;                \
            bf16x4 pk;                                                                  \
            pk[0] = (__bf16)src[0];                                                     \
            pk[1] = (__bf16)src[NLP];                                                   \
            pk[2] = (__bf16)src[2 * NLP];                                               \
            pk[3] = (__bf16)src[3 * NLP];                                               \
            *(bf16x4*)&Xs[spp][skk] = pk;                                               \
        }                                                                               \
        __syncthreads();                                                                \
        bf16x16 bm0 = cat8(ld8(&Xs[m][hi * 8]),      ld8(&Xs[m][hi * 8 + 16]));        \
        bf16x16 bm1 = cat8(ld8(&Xs[16 + m][hi * 8]), ld8(&Xs[16 + m][hi * 8 + 16]));   \
        ACCLIST                                                                         \
    }

template<int ACT, bool HAS_RES>
__global__ void __launch_bounds__(256)
conv1x1_wmma_kernel(const float* __restrict__ X, const __bf16* __restrict__ Wb,
                    const float* __restrict__ bias, const float* __restrict__ resid,
                    float* __restrict__ Y) {
    __shared__ __attribute__((aligned(16))) __bf16 Xs[32][40];
    const int t = threadIdx.x;
    const int wave = t >> 5, lane = t & 31;
    const int m = lane & 15, hi = lane >> 4;
    const int b  = blockIdx.x / PTILES;
    const int p0 = (blockIdx.x % PTILES) * 32;
    const float* xb = X + (size_t)b * BSTRIDE;
    const int spp = t & 31, skk = (t >> 5) << 2;
    const int o0 = wave * 16;
    f32x8 acc0 = zero8(), acc1 = zero8();
    CONV_MAINLOOP(
        const __bf16* wr = Wb + (size_t)(o0 + m) * DD + k0 + hi * 8;
        bf16x16 a = cat8(ld8(wr), ld8(wr + 16));
        acc0 = wmma_bf16(a, bm0, acc0);
        acc1 = wmma_bf16(a, bm1, acc1);
    )
    #pragma unroll
    for (int sub = 0; sub < 2; ++sub) {
        f32x8 acc = sub ? acc1 : acc0;
        const int col = p0 + sub * 16 + m;
        #pragma unroll
        for (int r = 0; r < 8; ++r) {
            const int oc = o0 + hi * 8 + r;
            float v = acc[r] + bias[oc];
            if (ACT == 1) v = fmaxf(v, 0.0f);
            size_t idx = (size_t)(b * DD + oc) * NLP + col;
            if (HAS_RES) v += resid[idx];
            Y[idx] = v;
        }
    }
}

// ---- dual conv + GLU epilogue:  Y = (W1@X+b1) * sigmoid(W2@X+b2) -----------
__global__ void __launch_bounds__(256)
conv1x1_glu_kernel(const float* __restrict__ X,
                   const __bf16* __restrict__ W1b, const float* __restrict__ b1,
                   const __bf16* __restrict__ W2b, const float* __restrict__ b2,
                   float* __restrict__ Y) {
    __shared__ __attribute__((aligned(16))) __bf16 Xs[32][40];
    const int t = threadIdx.x;
    const int wave = t >> 5, lane = t & 31;
    const int m = lane & 15, hi = lane >> 4;
    const int b  = blockIdx.x / PTILES;
    const int p0 = (blockIdx.x % PTILES) * 32;
    const float* xb = X + (size_t)b * BSTRIDE;
    const int spp = t & 31, skk = (t >> 5) << 2;
    const int o0 = wave * 16;
    f32x8 aA0 = zero8(), aA1 = zero8(), aB0 = zero8(), aB1 = zero8();
    CONV_MAINLOOP(
        const __bf16* w1 = W1b + (size_t)(o0 + m) * DD + k0 + hi * 8;
        const __bf16* w2 = W2b + (size_t)(o0 + m) * DD + k0 + hi * 8;
        bf16x16 a1 = cat8(ld8(w1), ld8(w1 + 16));
        bf16x16 a2 = cat8(ld8(w2), ld8(w2 + 16));
        aA0 = wmma_bf16(a1, bm0, aA0);
        aA1 = wmma_bf16(a1, bm1, aA1);
        aB0 = wmma_bf16(a2, bm0, aB0);
        aB1 = wmma_bf16(a2, bm1, aB1);
    )
    #pragma unroll
    for (int sub = 0; sub < 2; ++sub) {
        f32x8 accA = sub ? aA1 : aA0;
        f32x8 accB = sub ? aB1 : aB0;
        const int col = p0 + sub * 16 + m;
        #pragma unroll
        for (int r = 0; r < 8; ++r) {
            const int oc = o0 + hi * 8 + r;
            float x1 = accA[r] + b1[oc];
            float x2 = accB[r] + b2[oc];
            size_t idx = (size_t)(b * DD + oc) * NLP + col;
            Y[idx] = x1 * sigmoidf_(x2);
        }
    }
}

// ---- dual conv q/v + relu + head-softmax(q) + attention relayout -----------
// wave w == head h (16 output channels == DK). Writes:
//   Qs[b][h][l][n][x]  bf16 (softmax(relu(q)*0.25) over x)
//   Vt[b][h][l][y][n]  bf16 (relu(v))
__global__ void __launch_bounds__(256)
conv1x1_qv_kernel(const float* __restrict__ X,
                  const __bf16* __restrict__ Wqb, const float* __restrict__ qbias,
                  const __bf16* __restrict__ Wvb, const float* __restrict__ vbias,
                  __bf16* __restrict__ Qs, __bf16* __restrict__ Vt) {
    __shared__ __attribute__((aligned(16))) __bf16 Xs[32][40];
    const int t = threadIdx.x;
    const int wave = t >> 5, lane = t & 31;
    const int m = lane & 15, hi = lane >> 4;
    const int b  = blockIdx.x / PTILES;
    const int p0 = (blockIdx.x % PTILES) * 32;
    const float* xb = X + (size_t)b * BSTRIDE;
    const int spp = t & 31, skk = (t >> 5) << 2;
    const int o0 = wave * 16;
    f32x8 aQ0 = zero8(), aQ1 = zero8(), aV0 = zero8(), aV1 = zero8();
    CONV_MAINLOOP(
        const __bf16* wq = Wqb + (size_t)(o0 + m) * DD + k0 + hi * 8;
        const __bf16* wv = Wvb + (size_t)(o0 + m) * DD + k0 + hi * 8;
        bf16x16 a1 = cat8(ld8(wq), ld8(wq + 16));
        bf16x16 a2 = cat8(ld8(wv), ld8(wv + 16));
        aQ0 = wmma_bf16(a1, bm0, aQ0);
        aQ1 = wmma_bf16(a1, bm1, aQ1);
        aV0 = wmma_bf16(a2, bm0, aV0);
        aV1 = wmma_bf16(a2, bm1, aV1);
    )
    const int h = wave;
    #pragma unroll
    for (int sub = 0; sub < 2; ++sub) {
        f32x8 accQ = sub ? aQ1 : aQ0;
        f32x8 accV = sub ? aV1 : aV0;
        const int p = p0 + sub * 16 + m;     // pixel within batch
        const int n = p / LL, l = p % LL;
        // q: relu, scale, softmax over x (= M dim: this lane holds x = hi*8 + r)
        float q[8];
        float mx = -3.4e38f;
        #pragma unroll
        for (int r = 0; r < 8; ++r) {
            int oc = o0 + hi * 8 + r;
            q[r] = fmaxf(accQ[r] + qbias[oc], 0.0f) * 0.25f;
            mx = fmaxf(mx, q[r]);
        }
        mx = fmaxf(mx, __shfl_xor(mx, 16, 32));   // combine the two x-halves
        float s = 0.f;
        #pragma unroll
        for (int r = 0; r < 8; ++r) { q[r] = __expf(q[r] - mx); s += q[r]; }
        s += __shfl_xor(s, 16, 32);
        float inv = 1.0f / s;
        bf16x8 qpk;
        #pragma unroll
        for (int r = 0; r < 8; ++r) qpk[r] = (__bf16)(q[r] * inv);
        __bf16* qd = Qs + ((((size_t)(b * HH + h) * LL + l) * NN + n) * DKK + hi * 8);
        *(bf16x8*)qd = qpk;
        // v: relu, relayout to [y][n]
        __bf16* vd = Vt + (((size_t)(b * HH + h) * LL + l) * DKK + hi * 8) * NN + n;
        #pragma unroll
        for (int r = 0; r < 8; ++r) {
            int oc = o0 + hi * 8 + r;
            vd[(size_t)r * NN] = (__bf16)fmaxf(accV[r] + vbias[oc], 0.0f);
        }
    }
}

// ---- kv[x][y] = sum_n KT[x][n] * Vt[y][n]  (WMMA over N, 4 waves/block) ----
__global__ void __launch_bounds__(128)
kv_wmma_kernel(const __bf16* __restrict__ KT, const __bf16* __restrict__ Vt,
               float* __restrict__ kv) {
    __shared__ float red[4][16][16];
    const int t = threadIdx.x, wave = t >> 5, lane = t & 31;
    const int m = lane & 15, hi = lane >> 4;
    const int bhl = blockIdx.x;
    const int hl = bhl % (HH * LL);
    const __bf16* Ab = KT + (size_t)hl * (DKK * NN);    // [x][n]
    const __bf16* Bv = Vt + (size_t)bhl * (DKK * NN);   // [y][n]
    f32x8 acc = zero8();
    const int nbeg = wave * 512;
    for (int n0 = nbeg; n0 < nbeg + 512; n0 += 32) {
        const __bf16* ar = Ab + (size_t)m * NN + n0 + hi * 8;    // A: K in {0-7,16-23}/{8-15,24-31}
        bf16x16 a = cat8(ld8(ar), ld8(ar + 16));
        const __bf16* br = Bv + (size_t)m * NN + n0 + hi * 8;    // B: K in {0-7,16-23}/{8-15,24-31}
        bf16x16 bm = cat8(ld8(br), ld8(br + 16));
        acc = wmma_bf16(a, bm, acc);
    }
    #pragma unroll
    for (int r = 0; r < 8; ++r) red[wave][hi * 8 + r][m] = acc[r];
    __syncthreads();
    for (int i = t; i < 256; i += 128) {
        int row = i >> 4, col = i & 15;
        kv[(size_t)bhl * 256 + i] =
            red[0][row][col] + red[1][row][col] + red[2][row][col] + red[3][row][col];
    }
}

// ---- o[n][y] = sum_x Qs[n][x] * kv[x][y], relayout to (B,D,N,L) ------------
__global__ void __launch_bounds__(256)
attn_o_kernel(const __bf16* __restrict__ Qs, const float* __restrict__ kv,
              float* __restrict__ o_s) {
    __shared__ float kvs[256];
    const int bhl = blockIdx.x >> 3;
    const int chunk = blockIdx.x & 7;
    const int t = threadIdx.x;
    kvs[t] = kv[(size_t)bhl * 256 + t];
    __syncthreads();
    const int n = chunk * 256 + t;
    const int b = bhl / (HH * LL);
    const int hl = bhl % (HH * LL);
    const int h = hl / LL, l = hl % LL;
    const __bf16* q = Qs + ((size_t)bhl * NN + n) * DKK;
    float qf[16];
    #pragma unroll
    for (int x = 0; x < 16; ++x) qf[x] = (float)q[x];
    float* ob = o_s + ((size_t)(b * DD + h * DKK) * NN + n) * LL + l;
    #pragma unroll
    for (int y = 0; y < 16; ++y) {
        float s = 0.f;
        #pragma unroll
        for (int x = 0; x < 16; ++x) s = fmaf(qf[x], kvs[x * 16 + y], s);
        ob[(size_t)y * NLP] = s;
    }
}

// ---- layernorm over (C,N) per (b,l): two-stage deterministic reduction -----
__global__ void __launch_bounds__(256)
ln_part_kernel(const float* __restrict__ X, float* __restrict__ parts) {
    const int g = blockIdx.x >> 6, slab = blockIdx.x & 63;
    const int b = g / LL, l = g % LL;
    const float* xb = X + (size_t)b * BSTRIDE + l;
    float s = 0.f, ss = 0.f;
    for (int i = threadIdx.x; i < 4096; i += 256) {
        int c = i >> 5, nn = (i & 31) + slab * 32;
        float v = xb[((size_t)c * NN + nn) * LL];
        s += v; ss += v * v;
    }
    __shared__ float rs[256], rq[256];
    const int t = threadIdx.x;
    rs[t] = s; rq[t] = ss;
    __syncthreads();
    for (int o = 128; o > 0; o >>= 1) {
        if (t < o) { rs[t] += rs[t + o]; rq[t] += rq[t + o]; }
        __syncthreads();
    }
    if (t == 0) { parts[blockIdx.x * 2] = rs[0]; parts[blockIdx.x * 2 + 1] = rq[0]; }
}

__global__ void __launch_bounds__(64)
ln_stats_kernel(const float* __restrict__ parts, float* __restrict__ stats) {
    __shared__ float rs[64], rq[64];
    const int g = blockIdx.x, t = threadIdx.x;
    rs[t] = parts[(g * 64 + t) * 2];
    rq[t] = parts[(g * 64 + t) * 2 + 1];
    __syncthreads();
    for (int o = 32; o > 0; o >>= 1) {
        if (t < o) { rs[t] += rs[t + o]; rq[t] += rq[t + o]; }
        __syncthreads();
    }
    if (t == 0) {
        const float inv = 1.0f / (float)GRPSZ;
        float mean = rs[0] * inv;
        float var  = rq[0] * inv - mean * mean;
        stats[g * 2]     = mean;
        stats[g * 2 + 1] = rsqrtf(var + 1e-5f);
    }
}

__global__ void __launch_bounds__(256)
ln_apply_kernel(float* __restrict__ X, const float* __restrict__ stats,
                const float* __restrict__ sw, const float* __restrict__ sb) {
    const size_t stride = (size_t)gridDim.x * blockDim.x;
    for (size_t i = (size_t)blockIdx.x * blockDim.x + threadIdx.x; i < (size_t)TOT; i += stride) {
        int l = (int)(i % LL);
        int b = (int)(i / ((size_t)DD * NN * LL));
        int g = b * LL + l;
        size_t si = i % ((size_t)DD * NN * LL);   // index into (D,N,L) banks
        float mean = stats[g * 2], rstd = stats[g * 2 + 1];
        X[i] = (X[i] - mean) * rstd * sw[si] + sb[si];
    }
}

// ---- launcher --------------------------------------------------------------
extern "C" void kernel_launch(void* const* d_in, const int* in_sizes, int n_in,
                              void* d_out, int out_size, void* d_ws, size_t ws_size,
                              hipStream_t stream) {
    const float* h      = (const float*)d_in[0];
    const float* qW     = (const float*)d_in[1];
    const float* qb     = (const float*)d_in[2];
    const float* vW     = (const float*)d_in[3];
    const float* vb     = (const float*)d_in[4];
    const float* catW   = (const float*)d_in[5];
    const float* catb   = (const float*)d_in[6];
    const float* s_attn = (const float*)d_in[7];
    const float* g1W    = (const float*)d_in[8];
    const float* g1b    = (const float*)d_in[9];
    const float* g2W    = (const float*)d_in[10];
    const float* g2b    = (const float*)d_in[11];
    const float* g3W    = (const float*)d_in[12];
    const float* g3b    = (const float*)d_in[13];
    const float* fcW    = (const float*)d_in[14];
    const float* fcb    = (const float*)d_in[15];
    const float* s1     = (const float*)d_in[16];
    const float* s1b    = (const float*)d_in[17];
    const float* s2     = (const float*)d_in[18];
    const float* s2b    = (const float*)d_in[19];
    (void)in_sizes; (void)n_in; (void)out_size; (void)ws_size;

    char* ws = (char*)d_ws;
    size_t off = 0;
    auto alloc = [&](size_t bytes) -> void* {
        void* p = ws + off;
        off += (bytes + 255) & ~(size_t)255;
        return p;
    };

    __bf16* Wq  = (__bf16*)alloc((size_t)DD * DD * 2);
    __bf16* Wv  = (__bf16*)alloc((size_t)DD * DD * 2);
    __bf16* Wc  = (__bf16*)alloc((size_t)DD * DD * 2);
    __bf16* W1  = (__bf16*)alloc((size_t)DD * DD * 2);
    __bf16* W2  = (__bf16*)alloc((size_t)DD * DD * 2);
    __bf16* W3  = (__bf16*)alloc((size_t)DD * DD * 2);
    __bf16* Wf  = (__bf16*)alloc((size_t)DD * DD * 2);
    __bf16* KT  = (__bf16*)alloc((size_t)HH * LL * DKK * NN * 2);
    __bf16* Qs  = (__bf16*)alloc((size_t)TOT * 2);
    __bf16* Vt  = (__bf16*)alloc((size_t)TOT * 2);
    float*  kvb = (float*)alloc((size_t)BB * HH * LL * 256 * 4);
    float*  Ab  = (float*)alloc((size_t)TOT * 4);     // h_s chain
    float*  Bb  = (float*)alloc((size_t)TOT * 4);     // o_s / glu / h_f chain
    float*  parts = (float*)alloc((size_t)GRP * 64 * 2 * 4);
    float*  stats = (float*)alloc((size_t)GRP * 2 * 4);
    float*  out = (float*)d_out;

    const int wn = DD * DD;
    const int wblk = (wn + 255) / 256;
    cvt_bf16_kernel<<<wblk, 256, 0, stream>>>(qW,   Wq, wn);
    cvt_bf16_kernel<<<wblk, 256, 0, stream>>>(vW,   Wv, wn);
    cvt_bf16_kernel<<<wblk, 256, 0, stream>>>(catW, Wc, wn);
    cvt_bf16_kernel<<<wblk, 256, 0, stream>>>(g1W,  W1, wn);
    cvt_bf16_kernel<<<wblk, 256, 0, stream>>>(g2W,  W2, wn);
    cvt_bf16_kernel<<<wblk, 256, 0, stream>>>(g3W,  W3, wn);
    cvt_bf16_kernel<<<wblk, 256, 0, stream>>>(fcW,  Wf, wn);

    key_softmax_kernel<<<(HH * LL * NN + 255) / 256, 256, 0, stream>>>(s_attn, KT);

    // q,v conv + relu + softmax + relayout (h read once for both)
    conv1x1_qv_kernel<<<BB * PTILES, 256, 0, stream>>>(h, Wq, qb, Wv, vb, Qs, Vt);

    // kv per (b,h,l)
    kv_wmma_kernel<<<BB * HH * LL, 128, 0, stream>>>(KT, Vt, kvb);

    // o = Q @ kv  ->  (B,D,N,L) layout into Bb
    attn_o_kernel<<<BB * HH * LL * 8, 256, 0, stream>>>(Qs, kvb, Bb);

    // attn = relu(cat conv(o)) ; h_s_pre = h + attn  -> Ab
    conv1x1_wmma_kernel<1, true><<<BB * PTILES, 256, 0, stream>>>(Bb, Wc, catb, h, Ab);

    // layernorm(Ab) * s1 + s1b   (in place)
    ln_part_kernel<<<GRP * 64, 256, 0, stream>>>(Ab, parts);
    ln_stats_kernel<<<GRP, 64, 0, stream>>>(parts, stats);
    ln_apply_kernel<<<4096, 256, 0, stream>>>(Ab, stats, s1, s1b);

    // glu input g = (g1@h_s+b1) * sigmoid(g2@h_s+b2) -> Bb
    conv1x1_glu_kernel<<<BB * PTILES, 256, 0, stream>>>(Ab, W1, g1b, W2, g2b, Bb);

    // h_f_pre = h_s + g3@g  (in-place on Bb is safe: each block only touches
    // its own pixel columns, all reads precede the epilogue writes)
    conv1x1_wmma_kernel<0, true><<<BB * PTILES, 256, 0, stream>>>(Bb, W3, g3b, Ab, Bb);

    // layernorm(Bb) * s2 + s2b  (in place)
    ln_part_kernel<<<GRP * 64, 256, 0, stream>>>(Bb, parts);
    ln_stats_kernel<<<GRP, 64, 0, stream>>>(parts, stats);
    ln_apply_kernel<<<4096, 256, 0, stream>>>(Bb, stats, s2, s2b);

    // out = h_f + fc@h
    conv1x1_wmma_kernel<0, true><<<BB * PTILES, 256, 0, stream>>>(h, Wf, fcb, Bb, out);
}
